// GraphTransformerBackbone_87892210745839
// MI455X (gfx1250) — matrix operand order repaired and checked by reading, not compiled, not run on hardware
//
#include <hip/hip_runtime.h>
#include <hip/hip_bf16.h>
#include <math.h>

// Problem dims (fixed by reference)
#define BQ   8
#define NQ   512
#define NCC  256
#define DD   512
#define DCC  256
#define LL   12
#define NHH  8
#define DKK  64

typedef __attribute__((ext_vector_type(16))) __bf16 v16bf;
typedef __attribute__((ext_vector_type(8)))  float  v8f;

#if defined(__has_builtin)
#if __has_builtin(__builtin_amdgcn_tensor_load_to_lds) && \
    __has_builtin(__builtin_amdgcn_s_wait_tensorcnt)
#define HAVE_TDM 1
#endif
#endif
#ifndef HAVE_TDM
#define HAVE_TDM 0
#endif

__device__ __forceinline__ __bf16 f2bf(float f) { return (__bf16)f; }

__device__ __forceinline__ v8f v8f_zero() {
  v8f z = {0.f, 0.f, 0.f, 0.f, 0.f, 0.f, 0.f, 0.f};
  return z;
}

__device__ __forceinline__ v8f wmma_bf16(v16bf a, v16bf b, v8f c) {
  // D = A(16x32 bf16) * B(32x16 bf16) + C(16x16 f32)
  return __builtin_amdgcn_wmma_f32_16x16x32_bf16(false, a, false, b, (short)0, c,
                                                 false, false);
}

__device__ __forceinline__ float gelu_exact(float x) {
  return 0.5f * x * (1.0f + erff(x * 0.70710678118654752f));
}

// A-matrix (16x32 bf16) K index for element j of the v16bf (contiguous):
// lanes 0-15: K = j; lanes 16-31: K = 16 + j
__device__ __forceinline__ int a_kmap(int lane, int j) {
  int base = (lane & 16) ? 16 : 0;
  return base + j;
}

#if HAVE_TDM
typedef __attribute__((ext_vector_type(4))) unsigned int v4u;
typedef __attribute__((ext_vector_type(8))) int v8i;
typedef __attribute__((ext_vector_type(4))) int v4i;

// 2D TDM load: tile_h x tile_w bf16 elements, row stride stride_elems, into
// LDS at lds_addr with hardware row padding (pad codes per D# group1).
__device__ __forceinline__ void tdm_load_2d(const void* gsrc, unsigned lds_addr,
                                            unsigned tile_w, unsigned tile_h,
                                            unsigned stride_elems,
                                            unsigned pad_int_code,
                                            unsigned pad_amt_code) {
  unsigned long long ga = (unsigned long long)gsrc;
  v4u g0;
  g0.x = 1u;                                            // count=1 (valid), user
  g0.y = lds_addr;                                      // LDS byte address
  g0.z = (unsigned)ga;                                  // global_addr[31:0]
  g0.w = (unsigned)((ga >> 32) & 0x01ffffffu) | (2u << 30);  // [56:32] | type=2
  v8i g1;
  unsigned w0 = (1u << 16)                 // data_size = 1 -> 2 bytes
              | (1u << 20)                 // pad_enable
              | ((pad_int_code & 7u) << 22)
              | ((pad_amt_code & 127u) << 25);
  g1[0] = (int)w0;                         // wg_mask=0 (not in cluster)
  g1[1] = (int)((tile_w & 0xffffu) << 16); // tensor_dim0[15:0] @ bits 63:48
  g1[2] = (int)(((tile_w >> 16) & 0xffffu) | ((tile_h & 0xffffu) << 16));
  g1[3] = (int)(((tile_h >> 16) & 0xffffu) | ((tile_w & 0xffffu) << 16)); // tile_dim0
  g1[4] = (int)(tile_h & 0xffffu);         // tile_dim1 ; tile_dim2 = 0
  g1[5] = (int)stride_elems;               // tensor_dim0_stride[31:0]
  g1[6] = 0;
  g1[7] = 0;
  v4i z4 = {0, 0, 0, 0};
#if __clang_major__ >= 23
  v8i z8 = {0, 0, 0, 0, 0, 0, 0, 0};
  __builtin_amdgcn_tensor_load_to_lds(g0, g1, z4, z4, z8, 0);
#else
  __builtin_amdgcn_tensor_load_to_lds(g0, g1, z4, z4, 0);
#endif
}
#endif  // HAVE_TDM

// ---------------------------------------------------------------------------
// One-time fp32 -> bf16 conversion (weights / context), done once per launch.
// ---------------------------------------------------------------------------
__global__ void cvt_bf16_kernel(const float* __restrict__ x,
                                __bf16* __restrict__ y, size_t n) {
  size_t i = (size_t)blockIdx.x * blockDim.x + threadIdx.x;
  size_t stride = (size_t)gridDim.x * blockDim.x;
  for (; i < n; i += stride) y[i] = f2bf(x[i]);
}

// ---------------------------------------------------------------------------
// GEMM: C[M,N] = act(A[M,K]bf16 @ B[K,N]bf16 + bias[N]) (+ res[M,N])
// Block = 256 thr (8 waves), tile 128x128, K-step 32, double-buffered LDS.
// Staging: TDM tensor_load_to_lds DMA issued by wave 0 (TENSORcnt-gated),
// with hardware row padding; fallback to manual VMEM->LDS staging.
// Each wave owns a 32x64 subtile -> 8 WMMA accumulators.
// ---------------------------------------------------------------------------
template <bool GELU, bool OUTBF>
__global__ __launch_bounds__(256) void gemm_kernel(
    const __bf16* __restrict__ A, int lda, const __bf16* __restrict__ Bw,
    int ldb, const float* __restrict__ bias, const float* __restrict__ res,
    void* __restrict__ Cout, int ldc, int K) {
  // A rows: 16 DWORDs + 2 pad -> 36 bf16; B rows: 64 DWORDs + 2 pad -> 132.
  __shared__ __bf16 As[2][128][36];
  __shared__ __bf16 Bs[2][32][132];

  const int tid  = threadIdx.x;
  const int lane = tid & 31;
  const int wid  = tid >> 5;
  const int m0   = blockIdx.y * 128;
  const int n0   = blockIdx.x * 128;
  const int wr   = (wid >> 1) * 32;  // 4 row groups
  const int wc   = (wid & 1) * 64;   // 2 col groups
  const int l15  = lane & 15;
  const int bb0  = (lane & 16) ? 16 : 0;

  v8f acc[8];
#pragma unroll
  for (int i = 0; i < 8; ++i) acc[i] = v8f_zero();

#if !HAVE_TDM
  const int ar = tid >> 1, ac = (tid & 1) * 16;   // A: 128 x 32
  const int br = tid >> 3, bc2 = (tid & 7) * 16;  // B: 32 x 128
#endif

  auto stage = [&](int buf, int k0) {
#if HAVE_TDM
    if (tid < 32) {  // wave 0 issues the DMAs (wave-level ops, EXEC ignored)
      tdm_load_2d(A + (size_t)m0 * lda + k0,
                  (unsigned)(size_t)&As[buf][0][0], 32u, 128u, (unsigned)lda,
                  3u /*16 DW*/, 1u /*2 DW pad*/);
      tdm_load_2d(Bw + (size_t)k0 * ldb + n0,
                  (unsigned)(size_t)&Bs[buf][0][0], 128u, 32u, (unsigned)ldb,
                  5u /*64 DW*/, 1u /*2 DW pad*/);
    }
#else
    {
      const __bf16* sa = A + (size_t)(m0 + ar) * lda + (k0 + ac);
#pragma unroll
      for (int j = 0; j < 16; ++j) As[buf][ar][ac + j] = sa[j];
      const __bf16* sb = Bw + (size_t)(k0 + br) * ldb + (n0 + bc2);
#pragma unroll
      for (int j = 0; j < 16; ++j) Bs[buf][br][bc2 + j] = sb[j];
    }
#endif
  };

  stage(0, 0);
  for (int k0 = 0; k0 < K; k0 += 32) {
    const int cur = (k0 >> 5) & 1;
    const bool hasNext = (k0 + 32 < K);
    if (hasNext) stage(cur ^ 1, k0 + 32);
#if HAVE_TDM
    if (tid < 32) {
      // TDM ops retire in order: <=2 outstanding means current tile's
      // two DMAs are done (the two in flight belong to the next tile).
      if (hasNext) __builtin_amdgcn_s_wait_tensorcnt(2);
      else         __builtin_amdgcn_s_wait_tensorcnt(0);
    }
#endif
    __syncthreads();

    v16bf af[2], bfr[4];
#pragma unroll
    for (int s = 0; s < 2; ++s) {
      int am = wr + s * 16 + l15;
#pragma unroll
      for (int j = 0; j < 16; ++j) af[s][j] = As[cur][am][a_kmap(lane, j)];
    }
#pragma unroll
    for (int t = 0; t < 4; ++t) {
      int bc = wc + t * 16 + l15;
#pragma unroll
      for (int j = 0; j < 16; ++j) bfr[t][j] = Bs[cur][bb0 + j][bc];
    }
#pragma unroll
    for (int s = 0; s < 2; ++s)
#pragma unroll
      for (int t = 0; t < 4; ++t)
        acc[s * 4 + t] = wmma_bf16(af[s], bfr[t], acc[s * 4 + t]);
    __syncthreads();
  }

  const int rowadd = (lane & 16) ? 8 : 0;
#pragma unroll
  for (int s = 0; s < 2; ++s) {
#pragma unroll
    for (int t = 0; t < 4; ++t) {
      v8f a = acc[s * 4 + t];
      int col = n0 + wc + t * 16 + l15;
      float bv = bias[col];
#pragma unroll
      for (int r = 0; r < 8; ++r) {
        int row = m0 + wr + s * 16 + r + rowadd;
        float v = a[r] + bv;
        if (GELU) v = gelu_exact(v);
        if (OUTBF) {
          ((__bf16*)Cout)[(size_t)row * ldc + col] = f2bf(v);
        } else {
          if (res) v += res[(size_t)row * ldc + col];
          ((float*)Cout)[(size_t)row * ldc + col] = v;
        }
      }
    }
  }
}

// ---------------------------------------------------------------------------
// Fused attention: one wave handles (b, head, 16 query rows). bf16 Q/K/V.
// Q(16xDK) @ K^T -> scores (scale + optional bias + mask) -> softmax -> @V.
// qbuf rows are [B, NQ, q_ld]; kv rows are [B, NKEYS, kv_ld].
// ---------------------------------------------------------------------------
template <int NKEYS, bool HAS_BIAS>
__global__ __launch_bounds__(32) void attn_kernel(
    const __bf16* __restrict__ qbuf, int q_ld, int q_col_base,
    const __bf16* __restrict__ kvbuf, int kv_ld, int k_col_base,
    int v_col_base, const float* __restrict__ bias,
    const unsigned char* __restrict__ mask, __bf16* __restrict__ out,
    int out_ld, float scale) {
  __shared__ float  sc[16][NKEYS + 4];
  __shared__ __bf16 qs[16][66];

  const int lane = threadIdx.x;
  const int n0   = blockIdx.x * 16;
  const int hh   = blockIdx.y;
  const int b    = blockIdx.z;

  const __bf16* qb  = qbuf + (size_t)b * NQ * q_ld;
  const __bf16* kvb = kvbuf + (size_t)b * NKEYS * kv_ld;
  const int qc = q_col_base + hh * DKK;
  const int kc = k_col_base + hh * DKK;
  const int vc = v_col_base + hh * DKK;

  for (int idx = lane; idx < 16 * 64; idx += 32) {
    int m = idx >> 6, d = idx & 63;
    qs[m][d] = qb[(size_t)(n0 + m) * q_ld + qc + d];
  }
  __syncthreads();

  const int l15    = lane & 15;
  const int bb0    = (lane & 16) ? 16 : 0;
  const int rowadd = (lane & 16) ? 8 : 0;

  // ---- scores = Q @ K^T ----
  for (int kt = 0; kt < NKEYS / 32; ++kt) {
#pragma unroll
    for (int ch = 0; ch < 2; ++ch) {
      v8f acc = v8f_zero();
      int key = kt * 32 + ch * 16 + l15;
#pragma unroll
      for (int kk = 0; kk < 2; ++kk) {
        v16bf a, bv;
#pragma unroll
        for (int j = 0; j < 16; ++j) a[j] = qs[l15][32 * kk + a_kmap(lane, j)];
        const __bf16* kp = kvb + (size_t)key * kv_ld + kc + 32 * kk + bb0;
#pragma unroll
        for (int j = 0; j < 16; ++j) bv[j] = kp[j];
        acc = wmma_bf16(a, bv, acc);
      }
#pragma unroll
      for (int r = 0; r < 8; ++r) {
        int row  = r + rowadd;
        int keyc = kt * 32 + ch * 16 + l15;
        float v  = acc[r] * scale;
        if (HAS_BIAS)
          v += bias[(((size_t)b * NHH + hh) * NQ + (n0 + row)) * NKEYS + keyc];
        if (!mask[(size_t)b * NKEYS + keyc]) v = -1.0e9f;
        sc[row][keyc] = v;
      }
    }
  }
  __syncthreads();

  // ---- row softmax (2 lanes per row, each handles half the keys) ----
  {
    int row  = l15;
    int kbeg = (lane >> 4) * (NKEYS / 2);
    int kend = kbeg + NKEYS / 2;
    float mx = -3.0e38f;
    for (int k = kbeg; k < kend; ++k) mx = fmaxf(mx, sc[row][k]);
    mx = fmaxf(mx, __shfl_xor(mx, 16, 32));
    float s = 0.f;
    for (int k = kbeg; k < kend; ++k) {
      float e = __expf(sc[row][k] - mx);
      sc[row][k] = e;
      s += e;
    }
    s += __shfl_xor(s, 16, 32);
    float inv = 1.0f / s;
    for (int k = kbeg; k < kend; ++k) sc[row][k] *= inv;
  }
  __syncthreads();

  // ---- out = P @ V ----
  v8f acco[4];
#pragma unroll
  for (int c = 0; c < 4; ++c) acco[c] = v8f_zero();

  for (int kt = 0; kt < NKEYS / 32; ++kt) {
    v16bf a;
#pragma unroll
    for (int j = 0; j < 16; ++j)
      a[j] = f2bf(sc[l15][kt * 32 + a_kmap(lane, j)]);
#pragma unroll
    for (int c = 0; c < 4; ++c) {
      v16bf bv;
      const __bf16* vp = kvb + vc + c * 16 + l15;
#pragma unroll
      for (int j = 0; j < 16; ++j) {
        int key = kt * 32 + bb0 + j;
        bv[j] = vp[(size_t)key * kv_ld];
      }
      acco[c] = wmma_bf16(a, bv, acco[c]);
    }
  }

  __bf16* ob = out + (size_t)b * NQ * out_ld;
#pragma unroll
  for (int c = 0; c < 4; ++c)
#pragma unroll
    for (int r = 0; r < 8; ++r) {
      int row = n0 + r + rowadd;
      int col = hh * DKK + c * 16 + l15;
      ob[(size_t)row * out_ld + col] = f2bf(acco[c][r]);
    }
}

// ---------------------------------------------------------------------------
// LayerNorm: fp32 in, bf16 out (feeds WMMA GEMMs). One row per 128-thr block.
// ---------------------------------------------------------------------------
__global__ __launch_bounds__(128) void layernorm_kernel(
    const float* __restrict__ x, const float* __restrict__ g,
    const float* __restrict__ bt, __bf16* __restrict__ y) {
  __shared__ float s1[128], s2[128];
  const int row = blockIdx.x;
  const int tid = threadIdx.x;
  const float* xr = x + (size_t)row * DD;
  float a = 0.f, q = 0.f;
  for (int i = tid; i < DD; i += 128) {
    float v = xr[i];
    a += v;
    q += v * v;
  }
  s1[tid] = a;
  s2[tid] = q;
  __syncthreads();
  for (int st = 64; st > 0; st >>= 1) {
    if (tid < st) {
      s1[tid] += s1[tid + st];
      s2[tid] += s2[tid + st];
    }
    __syncthreads();
  }
  float mean = s1[0] * (1.0f / DD);
  float var  = s2[0] * (1.0f / DD) - mean * mean;
  float inv  = rsqrtf(var + 1e-5f);
  for (int i = tid; i < DD; i += 128)
    y[(size_t)row * DD + i] = f2bf((xr[i] - mean) * inv * g[i] + bt[i]);
}

// ---------------------------------------------------------------------------
// Time embedding MLP (per batch): sin/cos -> W1 -> gelu -> W2  (fp32, tiny)
// ---------------------------------------------------------------------------
__global__ __launch_bounds__(256) void time_embed_kernel(
    const int* __restrict__ t, const float* __restrict__ w1,
    const float* __restrict__ b1, const float* __restrict__ w2,
    const float* __restrict__ b2, float* __restrict__ temb) {
  __shared__ float emb[DD];
  __shared__ float h1[4 * DD];
  const int b   = blockIdx.x;
  const int tid = threadIdx.x;
  const float tv = (float)t[b];
  for (int i = tid; i < DD / 2; i += 256) {
    float f   = __expf(-logf(10000.f) * (float)i / (float)(DD / 2));
    float ang = tv * f;
    emb[i]          = __sinf(ang);
    emb[i + DD / 2] = __cosf(ang);
  }
  __syncthreads();
  for (int o = tid; o < 4 * DD; o += 256) {
    float acc = b1[o];
    for (int k = 0; k < DD; ++k) acc += emb[k] * w1[(size_t)k * 4 * DD + o];
    h1[o] = gelu_exact(acc);
  }
  __syncthreads();
  for (int o = tid; o < DD; o += 256) {
    float acc = b2[o];
    for (int k = 0; k < 4 * DD; ++k) acc += h1[k] * w2[(size_t)k * DD + o];
    temb[(size_t)b * DD + o] = acc;
  }
}

__global__ void degsum_kernel(const float* __restrict__ a_t,
                              float* __restrict__ deg) {
  int idx = blockIdx.x * blockDim.x + threadIdx.x;
  if (idx >= BQ * NQ) return;
  const float* r = a_t + (size_t)idx * NQ;
  float s = 0.f;
  for (int k = 0; k < NQ; ++k) s += r[k];
  deg[idx] = s;
}

__global__ void embed_kernel(const float* __restrict__ x_t,
                             const float* __restrict__ coord_w,
                             const float* __restrict__ coord_b,
                             const float* __restrict__ deg,
                             const float* __restrict__ deg_w,
                             const float* __restrict__ deg_b,
                             const float* __restrict__ temb,
                             const float* __restrict__ hdse_node,
                             float* __restrict__ h) {
  size_t idx = (size_t)blockIdx.x * blockDim.x + threadIdx.x;
  if (idx >= (size_t)BQ * NQ * DD) return;
  int d     = (int)(idx % DD);
  size_t rn = idx / DD;
  int b     = (int)(rn / NQ);
  float x0 = x_t[rn * 2], x1 = x_t[rn * 2 + 1];
  h[idx] = x0 * coord_w[d] + x1 * coord_w[DD + d] + coord_b[d] +
           deg[rn] * deg_w[d] + deg_b[d] + temb[(size_t)b * DD + d] +
           hdse_node[idx];
}

__global__ void eps_kernel(const __bf16* __restrict__ hn,
                           const float* __restrict__ ch_w,
                           const float* __restrict__ ch_b,
                           const unsigned char* __restrict__ nmask,
                           float* __restrict__ out) {
  int idx = blockIdx.x * blockDim.x + threadIdx.x;
  if (idx >= BQ * NQ * 2) return;
  int j  = idx & 1;
  int rn = idx >> 1;
  int b  = rn / NQ;
  int n  = rn % NQ;
  const __bf16* hr = hn + (size_t)rn * DD;
  float acc = ch_b[j];
  for (int k = 0; k < DD; ++k) acc += (float)hr[k] * ch_w[k * 2 + j];
  out[idx] = nmask[(size_t)b * NQ + n] ? acc : 0.f;
}

// el NT GEMM: G[b] = hs[b] (NQ x 128 bf16) @ hd[b]^T -> one wave / 32x32 tile
__global__ __launch_bounds__(32) void el_gemm_kernel(
    const __bf16* __restrict__ hs, const __bf16* __restrict__ hd,
    float* __restrict__ G) {
  const int lane = threadIdx.x;
  const int c0   = blockIdx.x * 32;
  const int r0   = blockIdx.y * 32;
  const int b    = blockIdx.z;
  const __bf16* hsb = hs + (size_t)b * NQ * 128;
  const __bf16* hdb = hd + (size_t)b * NQ * 128;
  const int l15 = lane & 15;
  const int bb0 = (lane & 16) ? 16 : 0;

  v8f acc[4];
#pragma unroll
  for (int i = 0; i < 4; ++i) acc[i] = v8f_zero();

  for (int k0 = 0; k0 < 128; k0 += 32) {
    v16bf a[2], bv[2];
#pragma unroll
    for (int s = 0; s < 2; ++s) {
      int row = r0 + s * 16 + l15;
#pragma unroll
      for (int j = 0; j < 16; ++j)
        a[s][j] = hsb[(size_t)row * 128 + k0 + a_kmap(lane, j)];
      int col = c0 + s * 16 + l15;
      const __bf16* hp = hdb + (size_t)col * 128 + k0 + bb0;
#pragma unroll
      for (int j = 0; j < 16; ++j) bv[s][j] = hp[j];
    }
    acc[0] = wmma_bf16(a[0], bv[0], acc[0]);
    acc[1] = wmma_bf16(a[0], bv[1], acc[1]);
    acc[2] = wmma_bf16(a[1], bv[0], acc[2]);
    acc[3] = wmma_bf16(a[1], bv[1], acc[3]);
  }

  const int rowadd = (lane & 16) ? 8 : 0;
  float* Gb = G + (size_t)b * NQ * NQ;
#pragma unroll
  for (int s = 0; s < 2; ++s)
#pragma unroll
    for (int t2 = 0; t2 < 2; ++t2)
#pragma unroll
      for (int r = 0; r < 8; ++r) {
        int row = r0 + s * 16 + r + rowadd;
        int col = c0 + t2 * 16 + l15;
        Gb[(size_t)row * NQ + col] = acc[s * 2 + t2][r];
      }
}

__global__ void el_post_kernel(const float* __restrict__ G,
                               const unsigned char* __restrict__ nmask,
                               const float* __restrict__ eb,
                               float* __restrict__ out) {
  size_t idx = (size_t)blockIdx.x * blockDim.x + threadIdx.x;
  if (idx >= (size_t)BQ * NQ * NQ) return;
  int m    = (int)(idx % NQ);
  size_t t = idx / NQ;
  int n    = (int)(t % NQ);
  int b    = (int)(t / NQ);
  float v = 0.f;
  if (n != m && nmask[(size_t)b * NQ + n] && nmask[(size_t)b * NQ + m])
    v = 0.5f * (G[idx] + G[((size_t)b * NQ + m) * NQ + n]) + eb[0];
  out[idx] = v;
}

// ---------------------------------------------------------------------------
extern "C" void kernel_launch(void* const* d_in, const int* in_sizes, int n_in,
                              void* d_out, int out_size, void* d_ws,
                              size_t ws_size, hipStream_t stream) {
  (void)in_sizes; (void)n_in; (void)out_size; (void)ws_size;

  const float* x_t       = (const float*)d_in[0];
  const float* a_t       = (const float*)d_in[1];
  const float* hdse_node = (const float*)d_in[2];
  const float* hdse_bias = (const float*)d_in[3];
  const float* context   = (const float*)d_in[4];
  const float* coord_w   = (const float*)d_in[5];
  const float* coord_b   = (const float*)d_in[6];
  const float* deg_w     = (const float*)d_in[7];
  const float* deg_b     = (const float*)d_in[8];
  const float* tm_w1     = (const float*)d_in[9];
  const float* tm_b1     = (const float*)d_in[10];
  const float* tm_w2     = (const float*)d_in[11];
  const float* tm_b2     = (const float*)d_in[12];
  const float* ln_sa_g   = (const float*)d_in[13];
  const float* ln_sa_b   = (const float*)d_in[14];
  const float* qkv_w     = (const float*)d_in[15];
  const float* qkv_b     = (const float*)d_in[16];
  const float* osa_w     = (const float*)d_in[17];
  const float* osa_b     = (const float*)d_in[18];
  const float* ln_ca_g   = (const float*)d_in[19];
  const float* ln_ca_b   = (const float*)d_in[20];
  const float* qca_w     = (const float*)d_in[21];
  const float* qca_b     = (const float*)d_in[22];
  const float* kvca_w    = (const float*)d_in[23];
  const float* kvca_b    = (const float*)d_in[24];
  const float* oca_w     = (const float*)d_in[25];
  const float* oca_b     = (const float*)d_in[26];
  const float* ln_ff_g   = (const float*)d_in[27];
  const float* ln_ff_b   = (const float*)d_in[28];
  const float* ff1_w     = (const float*)d_in[29];
  const float* ff1_b     = (const float*)d_in[30];
  const float* ff2_w     = (const float*)d_in[31];
  const float* ff2_b     = (const float*)d_in[32];
  const float* fn_g      = (const float*)d_in[33];
  const float* fn_b      = (const float*)d_in[34];
  const float* ch_w      = (const float*)d_in[35];
  const float* ch_b      = (const float*)d_in[36];
  const float* es_w      = (const float*)d_in[37];
  const float* es_b      = (const float*)d_in[38];
  const float* ed_w      = (const float*)d_in[39];
  const float* ed_b      = (const float*)d_in[40];
  const float* eb        = (const float*)d_in[41];
  const int*   t_in      = (const int*)d_in[42];
  const unsigned char* node_mask    = (const unsigned char*)d_in[43];
  const unsigned char* context_mask = (const unsigned char*)d_in[44];

  // ---- workspace carve-out ----
  char* wsp  = (char*)d_ws;
  size_t off = 0;
  auto alloc = [&](size_t bytes) -> void* {
    void* p = (void*)(wsp + off);
    off += (bytes + 255) & ~(size_t)255;
    return p;
  };
  // bf16 weight copies (converted once per launch)
  __bf16* qkv_wb  = (__bf16*)alloc((size_t)LL * DD * 3 * DD * 2);
  __bf16* osa_wb  = (__bf16*)alloc((size_t)LL * DD * DD * 2);
  __bf16* qca_wb  = (__bf16*)alloc((size_t)LL * DD * DD * 2);
  __bf16* kvca_wb = (__bf16*)alloc((size_t)LL * DCC * 2 * DD * 2);
  __bf16* oca_wb  = (__bf16*)alloc((size_t)LL * DD * DD * 2);
  __bf16* ff1_wb  = (__bf16*)alloc((size_t)LL * DD * 4 * DD * 2);
  __bf16* ff2_wb  = (__bf16*)alloc((size_t)LL * 4 * DD * DD * 2);
  __bf16* es_wb   = (__bf16*)alloc((size_t)DD * 128 * 2);
  __bf16* ed_wb   = (__bf16*)alloc((size_t)DD * 128 * 2);
  __bf16* ctx_b   = (__bf16*)alloc((size_t)BQ * NCC * DCC * 2);
  // activations
  float*  temb = (float*)alloc((size_t)BQ * DD * 4);
  float*  deg  = (float*)alloc((size_t)BQ * NQ * 4);
  float*  h    = (float*)alloc((size_t)BQ * NQ * DD * 4);      // residual fp32
  __bf16* hn   = (__bf16*)alloc((size_t)BQ * NQ * DD * 2);     // LN output
  __bf16* qkv  = (__bf16*)alloc((size_t)BQ * NQ * 3 * DD * 2); // also q_ca
  __bf16* attn = (__bf16*)alloc((size_t)BQ * NQ * DD * 2);
  __bf16* kvca = (__bf16*)alloc((size_t)BQ * NCC * 2 * DD * 2);
  __bf16* ff1  = (__bf16*)alloc((size_t)BQ * NQ * 4 * DD * 2);
  __bf16* hsb  = (__bf16*)alloc((size_t)BQ * NQ * 128 * 2);
  __bf16* hdb  = (__bf16*)alloc((size_t)BQ * NQ * 128 * 2);
  float*  elg  = (float*)alloc((size_t)BQ * NQ * NQ * 4);

  const int Mrows = BQ * NQ;  // 4096
  const float scale = 0.125f; // 1/sqrt(64)

  // ---- one-time weight conversions ----
  auto cvt = [&](const float* src, __bf16* dst, size_t n) {
    int blocks = (int)((n + 1023) / 1024);
    if (blocks > 16384) blocks = 16384;
    cvt_bf16_kernel<<<blocks, 256, 0, stream>>>(src, dst, n);
  };
  cvt(qkv_w, qkv_wb, (size_t)LL * DD * 3 * DD);
  cvt(osa_w, osa_wb, (size_t)LL * DD * DD);
  cvt(qca_w, qca_wb, (size_t)LL * DD * DD);
  cvt(kvca_w, kvca_wb, (size_t)LL * DCC * 2 * DD);
  cvt(oca_w, oca_wb, (size_t)LL * DD * DD);
  cvt(ff1_w, ff1_wb, (size_t)LL * DD * 4 * DD);
  cvt(ff2_w, ff2_wb, (size_t)LL * 4 * DD * DD);
  cvt(es_w, es_wb, (size_t)DD * 128);
  cvt(ed_w, ed_wb, (size_t)DD * 128);
  cvt(context, ctx_b, (size_t)BQ * NCC * DCC);

  // ---- embedding ----
  time_embed_kernel<<<BQ, 256, 0, stream>>>(t_in, tm_w1, tm_b1, tm_w2, tm_b2,
                                            temb);
  degsum_kernel<<<(BQ * NQ + 255) / 256, 256, 0, stream>>>(a_t, deg);
  embed_kernel<<<(int)(((size_t)Mrows * DD + 255) / 256), 256, 0, stream>>>(
      x_t, coord_w, coord_b, deg, deg_w, deg_b, temb, hdse_node, h);

  for (int i = 0; i < LL; ++i) {
    // ---- self attention ----
    layernorm_kernel<<<Mrows, 128, 0, stream>>>(h, ln_sa_g + i * DD,
                                                ln_sa_b + i * DD, hn);
    gemm_kernel<false, true><<<dim3(3 * DD / 128, Mrows / 128), 256, 0,
                               stream>>>(
        hn, DD, qkv_wb + (size_t)i * DD * 3 * DD, 3 * DD, qkv_b + i * 3 * DD,
        nullptr, qkv, 3 * DD, DD);
    attn_kernel<NQ, true><<<dim3(NQ / 16, NHH, BQ), 32, 0, stream>>>(
        qkv, 3 * DD, 0, qkv, 3 * DD, DD, 2 * DD, hdse_bias, node_mask, attn,
        DD, scale);
    gemm_kernel<false, false><<<dim3(DD / 128, Mrows / 128), 256, 0, stream>>>(
        attn, DD, osa_wb + (size_t)i * DD * DD, DD, osa_b + i * DD, h, h, DD,
        DD);

    // ---- cross attention ----
    layernorm_kernel<<<Mrows, 128, 0, stream>>>(h, ln_ca_g + i * DD,
                                                ln_ca_b + i * DD, hn);
    gemm_kernel<false, true><<<dim3(DD / 128, Mrows / 128), 256, 0, stream>>>(
        hn, DD, qca_wb + (size_t)i * DD * DD, DD, qca_b + i * DD, nullptr, qkv,
        DD, DD);
    gemm_kernel<false, true><<<dim3(2 * DD / 128, (BQ * NCC) / 128), 256, 0,
                               stream>>>(
        ctx_b, DCC, kvca_wb + (size_t)i * DCC * 2 * DD, 2 * DD,
        kvca_b + i * 2 * DD, nullptr, kvca, 2 * DD, DCC);
    attn_kernel<NCC, false><<<dim3(NQ / 16, NHH, BQ), 32, 0, stream>>>(
        qkv, DD, 0, kvca, 2 * DD, 0, DD, nullptr, context_mask, attn, DD,
        scale);
    gemm_kernel<false, false><<<dim3(DD / 128, Mrows / 128), 256, 0, stream>>>(
        attn, DD, oca_wb + (size_t)i * DD * DD, DD, oca_b + i * DD, h, h, DD,
        DD);

    // ---- FFN ----
    layernorm_kernel<<<Mrows, 128, 0, stream>>>(h, ln_ff_g + i * DD,
                                                ln_ff_b + i * DD, hn);
    gemm_kernel<true, true><<<dim3(4 * DD / 128, Mrows / 128), 256, 0,
                              stream>>>(
        hn, DD, ff1_wb + (size_t)i * DD * 4 * DD, 4 * DD, ff1_b + i * 4 * DD,
        nullptr, ff1, 4 * DD, DD);
    gemm_kernel<false, false><<<dim3(DD / 128, Mrows / 128), 256, 0, stream>>>(
        ff1, 4 * DD, ff2_wb + (size_t)i * 4 * DD * DD, DD, ff2_b + i * DD, h,
        h, DD, 4 * DD);
  }

  // ---- heads ----
  layernorm_kernel<<<Mrows, 128, 0, stream>>>(h, fn_g, fn_b, hn);
  eps_kernel<<<(BQ * NQ * 2 + 255) / 256, 256, 0, stream>>>(
      hn, ch_w, ch_b, node_mask, (float*)d_out);
  gemm_kernel<false, true><<<dim3(128 / 128, Mrows / 128), 256, 0, stream>>>(
      hn, DD, es_wb, 128, es_b, nullptr, hsb, 128, DD);
  gemm_kernel<false, true><<<dim3(128 / 128, Mrows / 128), 256, 0, stream>>>(
      hn, DD, ed_wb, 128, ed_b, nullptr, hdb, 128, DD);
  el_gemm_kernel<<<dim3(NQ / 32, NQ / 32, BQ), 32, 0, stream>>>(hsb, hdb, elg);
  el_post_kernel<<<(int)(((size_t)BQ * NQ * NQ + 255) / 256), 256, 0,
                   stream>>>(elg, node_mask, eb,
                             (float*)d_out + (size_t)BQ * NQ * 2);
}
